// UniPhyBlock_6519760355619
// MI455X (gfx1250) — matrix operand order repaired and checked by reading, not compiled, not run on hardware
//
#include <hip/hip_runtime.h>
#include <stdint.h>
#include <math.h>

// ---------------------------------------------------------------------------
// UniPhyBlock for MI455X (gfx1250, wave32, WMMA).
// All heavy contractions mapped to v_wmma_f32_16x16x32_f16.
// FFT2 implemented as DFT-matrix GEMMs (WMMA) instead of butterflies.
// Conv input zero-haloed -> branchless implicit GEMM (no exec churn).
// Spectral image staged to LDS via global_load_async_to_lds_b128 (ASYNCcnt).
// ---------------------------------------------------------------------------

#define DEVI __device__ __forceinline__

typedef __attribute__((ext_vector_type(16))) _Float16 v16h;
typedef __attribute__((ext_vector_type(8)))  float    v8f;
typedef _Float16 h16;

constexpr int BB = 2, TT = 16, CC = 64, HH = 64, WW = 64;
constexpr int BT = BB * TT;            // 32
constexpr int HW = HH * WW;            // 4096
constexpr int NPOS = BT * HW;          // 131072 (spatial positions)
constexpr int NPOST = BB * HW;         // 8192  (temporal scan positions)
constexpr int HP = 66;                 // padded image extent (1-halo)

// ---------------- workspace layout (bytes, 256B aligned) -------------------
constexpr size_t SZ_XN16P   = (size_t)BT * HP * HP * 128 * 2; // padded conv in
constexpr size_t SZ_XN16    = (size_t)NPOS * 128 * 2;         // f16 [pos][128]
constexpr size_t SZ_PLANE16 = (size_t)BT * CC * HW * 2;       // f16 [bt][c][hw]
constexpr size_t SZ_PLANE32 = (size_t)BT * CC * HW * 4;       // f32 [bt][c][hw]
constexpr size_t SZ_CONVO   = (size_t)NPOS * 128 * 4;         // f32 [pos][128]

constexpr size_t OFF_XN16P = 0;
constexpr size_t OFF_XNR   = OFF_XN16P + SZ_XN16P;
constexpr size_t OFF_XNI   = OFF_XNR   + SZ_PLANE16;
constexpr size_t OFF_CONVO = OFF_XNI   + SZ_PLANE16;
constexpr size_t OFF_ZR    = OFF_CONVO + SZ_CONVO;
constexpr size_t OFF_ZI    = OFF_ZR    + SZ_PLANE32;
constexpr size_t OFF_ZN16  = OFF_ZI    + SZ_PLANE32;
constexpr size_t OFF_XBR   = OFF_ZN16  + SZ_XN16;
constexpr size_t OFF_XBI   = OFF_XBR   + SZ_PLANE32;
constexpr size_t OFF_LNM16 = OFF_XBI   + SZ_PLANE32;
constexpr size_t OFF_CONVW = OFF_LNM16 + SZ_XN16;             // f16 [128][1152]
constexpr size_t OFF_WIN   = OFF_CONVW + (size_t)147456 * 2;  // f16 [1024][128]
constexpr size_t OFF_WOUT  = OFF_WIN   + (size_t)131072 * 2;  // f16 [128][512]
constexpr size_t OFF_BPK   = OFF_WOUT  + (size_t)65536 * 2;   // f16 3x[64][64]
constexpr size_t OFF_CPK   = OFF_BPK   + (size_t)3 * 4096 * 2;
constexpr size_t OFF_FPK   = OFF_CPK   + (size_t)3 * 4096 * 2;
constexpr size_t OFF_FIPK  = OFF_FPK   + (size_t)3 * 4096 * 2;
constexpr size_t OFF_ATAB  = OFF_FIPK  + (size_t)3 * 4096 * 2; // f32 [BT][C]x2

// ---------------- small helpers --------------------------------------------
DEVI v8f zero8() {
    v8f z;
#pragma unroll
    for (int i = 0; i < 8; ++i) z[i] = 0.0f;
    return z;
}

// WMMA wrapper: D = A(16x32 f16) * B(32x16 f16) + C(16x16 f32)
DEVI v8f wmma16(v16h a, v16h b, v8f c) {
    return __builtin_amdgcn_wmma_f32_16x16x32_f16(
        /*neg_a=*/false, a, /*neg_b=*/false, b,
        /*c_mod=*/(short)0, c, /*reuse_a=*/false, /*reuse_b=*/false);
}

// A fragment: row-major [row][K], base p points at (row0, k0), stride ldk.
// lane<16 -> M=lane, K {0..7,16..23}; lane>=16 -> M=lane-16, K {8..15,24..31}
DEVI v16h load_a_frag(const h16* p, int ldk) {
    const int lane = threadIdx.x & 31;
    const h16* r = p + (size_t)(lane & 15) * ldk + ((lane >> 4) << 3);
    v16h a;
    ((uint4*)&a)[0] = *(const uint4*)(r);
    ((uint4*)&a)[1] = *(const uint4*)(r + 16);
    return a;
}

// B fragment from n-major pack [N][K]: lane -> N = lane&15,
// lane<16 -> K {0..7,16..23}; lane>=16 -> K {8..15,24..31}
DEVI v16h load_b_frag(const h16* p, int ldk) {
    const int lane = threadIdx.x & 31;
    const h16* r = p + (size_t)(lane & 15) * ldk + ((lane >> 4) << 3);
    v16h b;
    ((uint4*)&b)[0] = *(const uint4*)(r);
    ((uint4*)&b)[1] = *(const uint4*)(r + 16);
    return b;
}

// complex GEMM K=64 with pre-hoisted A fragments; B pack = {r, i, -i} planes
DEVI void cgemm_pre(v16h ar0, v16h ai0, v16h ar1, v16h ai1,
                    const h16* Bp, int n0, v8f& cr, v8f& ci) {
    v16h br  = load_b_frag(Bp +        n0 * 64, 64);
    v16h bi  = load_b_frag(Bp + 4096 + n0 * 64, 64);
    v16h bni = load_b_frag(Bp + 8192 + n0 * 64, 64);
    cr = wmma16(ar0, br,  cr);
    cr = wmma16(ai0, bni, cr);   // - Ai*Bi
    ci = wmma16(ar0, bi,  ci);
    ci = wmma16(ai0, br,  ci);
    br  = load_b_frag(Bp +        n0 * 64 + 32, 64);
    bi  = load_b_frag(Bp + 4096 + n0 * 64 + 32, 64);
    bni = load_b_frag(Bp + 8192 + n0 * 64 + 32, 64);
    cr = wmma16(ar1, br,  cr);
    cr = wmma16(ai1, bni, cr);
    ci = wmma16(ar1, bi,  ci);
    ci = wmma16(ai1, br,  ci);
}

DEVI float wave_sum(float v) {
#pragma unroll
    for (int i = 16; i > 0; i >>= 1) v += __shfl_xor(v, i, 32);
    return v;
}

DEVI uint32_t ihash(uint32_t x) {
    x ^= x >> 16; x *= 0x7feb352du;
    x ^= x >> 15; x *= 0x846ca68bu;
    x ^= x >> 16; return x;
}

// ===========================================================================
// K0: repack weights / DFT matrices / A_op table into WMMA-friendly f16
// ===========================================================================
__global__ void k_pack(const float* conv_w, const float* w_in, const float* w_out,
                       const float* bop_r, const float* bop_i,
                       const float* cop_r, const float* cop_i,
                       const float* alpha, const float* omega, const float* dt,
                       h16* convw, h16* win, h16* wout,
                       h16* bpk, h16* cpk, h16* fpk, h16* fipk, float* atab) {
    int gid = blockIdx.x * 256 + threadIdx.x;
    if (gid < 147456) {   // conv: [oc][tap*128+ci] <- conv_w[oc][ci][ky][kx]
        int oc = gid / 1152, r = gid % 1152, tap = r / 128, ci = r % 128;
        convw[gid] = (h16)conv_w[(size_t)oc * 1152 + (size_t)ci * 9 + tap];
        return;
    }
    gid -= 147456;
    if (gid < 131072) {   // w_in: [n=1024][k=128]
        int n = gid / 128, k = gid % 128;
        win[gid] = (h16)w_in[(size_t)k * 1024 + n];
        return;
    }
    gid -= 131072;
    if (gid < 65536) {    // w_out: [n=128][k=512]
        int n = gid / 512, k = gid % 512;
        wout[gid] = (h16)w_out[(size_t)k * 128 + n];
        return;
    }
    gid -= 65536;
    if (gid < 4096) {     // B_op pack {r, i, -i} (n-major == row-major)
        bpk[gid] = (h16)bop_r[gid];
        bpk[4096 + gid] = (h16)bop_i[gid];
        bpk[8192 + gid] = (h16)(-bop_i[gid]);
        return;
    }
    gid -= 4096;
    if (gid < 4096) {     // C_op pack
        cpk[gid] = (h16)cop_r[gid];
        cpk[4096 + gid] = (h16)cop_i[gid];
        cpk[8192 + gid] = (h16)(-cop_i[gid]);
        return;
    }
    gid -= 4096;
    if (gid < 4096) {     // DFT F (fwd) and Fi (inv, /64), symmetric matrices
        int j = gid / 64, k = gid % 64;
        int e = (j * k) & 63;
        float a = -6.28318530717958647692f * (float)e * (1.0f / 64.0f);
        float cr = cosf(a), ci = sinf(a);
        fpk[gid] = (h16)cr;  fpk[4096 + gid] = (h16)ci;  fpk[8192 + gid] = (h16)(-ci);
        float s = 1.0f / 64.0f;
        fipk[gid] = (h16)(cr * s); fipk[4096 + gid] = (h16)(-ci * s);
        fipk[8192 + gid] = (h16)(ci * s);
        return;
    }
    gid -= 4096;
    if (gid < 2048) {     // A_op[bt][c] = exp((-softplus(alpha)+i*omega)*dt)
        int bt = gid / 64, c = gid % 64;
        float d = dt[bt];
        float a = alpha[c];
        float sp = (a > 20.0f) ? a : log1pf(expf(a));
        float mag = expf(-sp * d);
        atab[2 * gid + 0] = mag * cosf(omega[c] * d);
        atab[2 * gid + 1] = mag * sinf(omega[c] * d);
    }
}

// ===========================================================================
// K1: spatial complex LayerNorm (wave per position, 128 features)
// writes zero-haloed xn16p [bt][66][66][128] and xnr/xni planes [bt][c][hw]
// ===========================================================================
__global__ void k_ln_spatial(const float* x_r, const float* x_i,
                             const float* g, const float* bta,
                             h16* xn16p, h16* xnr, h16* xni) {
    const int lane = threadIdx.x & 31;
    const int gpos = blockIdx.x * 8 + (threadIdx.x >> 5);
    const int bt = gpos >> 12, p = gpos & 4095;
    const int hh = p >> 6, ww = p & 63;
    float v[4];
#pragma unroll
    for (int j = 0; j < 4; ++j) {
        int f = lane + 32 * j;
        v[j] = (f < 64) ? x_r[(size_t)((bt << 6) + f) * 4096 + p]
                        : x_i[(size_t)((bt << 6) + f - 64) * 4096 + p];
    }
    float s = 0.f, s2 = 0.f;
#pragma unroll
    for (int j = 0; j < 4; ++j) { s += v[j]; s2 += v[j] * v[j]; }
    s = wave_sum(s); s2 = wave_sum(s2);
    float mu = s * (1.0f / 128.0f);
    float inv = rsqrtf(s2 * (1.0f / 128.0f) - mu * mu + 1e-5f);
    const size_t prow = ((size_t)((bt * HP + hh + 1) * HP + ww + 1)) << 7;
#pragma unroll
    for (int j = 0; j < 4; ++j) {
        int f = lane + 32 * j;
        float y = (v[j] - mu) * inv * g[f] + bta[f];
        xn16p[prow + f] = (h16)y;
        if (f < 64) xnr[(size_t)((bt << 6) + f) * 4096 + p] = (h16)y;
        else        xni[(size_t)((bt << 6) + f - 64) * 4096 + p] = (h16)y;
    }
}

// ===========================================================================
// K2: 3x3 conv as implicit GEMM on zero-haloed input (branchless A loads)
// K = 9*128 = 1152; block = one 16-position M-tile; wave = one 16-oc N-tile
// ===========================================================================
__global__ void k_conv(const h16* __restrict__ xn16p, const h16* __restrict__ convw,
                       const float* conv_b, float* __restrict__ convo) {
    const int lane = threadIdx.x & 31;
    const int n0 = (threadIdx.x >> 5) << 4;       // 0..112
    const int gbase = blockIdx.x * 16;            // first position of tile
    const int bt = gbase >> 12;
    const int p0 = gbase & 4095;
    const int h0 = p0 >> 6, w0 = p0 & 63;
    const int wl = w0 + (lane & 15);              // this lane's column position

    v8f acc = zero8();
#pragma unroll
    for (int tap = 0; tap < 9; ++tap) {
        const int dy = tap / 3, dx = tap % 3;     // halo absorbs the -1
        const int hy = h0 + dy, wx = wl + dx;     // indices into padded image
        const h16* ap0 = xn16p + ((size_t)((bt * HP + hy) * HP + wx) << 7)
                                + ((lane >> 4) << 3);
        __builtin_prefetch(convw + (size_t)n0 * 1152 + (tap + 1) * 128, 0, 0);
#pragma unroll
        for (int chunk = 0; chunk < 4; ++chunk) {
            const int kb = chunk << 5;
            v16h a;
            ((uint4*)&a)[0] = *(const uint4*)(ap0 + kb);
            ((uint4*)&a)[1] = *(const uint4*)(ap0 + kb + 16);
            v16h b = load_b_frag(convw + (size_t)n0 * 1152 + tap * 128 + kb, 1152);
            acc = wmma16(a, b, acc);
        }
    }
    const int oc = n0 + (lane & 15);
    const float bias = conv_b[oc];
#pragma unroll
    for (int vv = 0; vv < 8; ++vv) {
        int m = vv + ((lane >> 4) << 3);
        convo[((size_t)(gbase + m) << 7) + oc] = acc[vv] + bias;
    }
}

// ===========================================================================
// K3: spectral branch as DFT GEMMs, per (bt,c) image held in LDS (f16).
// Image staged memory->LDS with global_load_async_to_lds_b128 (ASYNCcnt).
// 4 complex GEMM stages with transposed inter-stage stores; wspec twiddle
// folded into stage 2; gate combine + residual folded into stage 4.
// ===========================================================================
__global__ void k_spec(const h16* __restrict__ xnr, const h16* __restrict__ xni,
                       const h16* __restrict__ fpk, const h16* __restrict__ fipk,
                       const float* spec_wr, const float* spec_wi,
                       const float* convo, const float* x_r, const float* x_i,
                       const float* gatep, float* zr, float* zi) {
    __shared__ h16 sAr[2][4096];
    __shared__ h16 sAi[2][4096];
    const int lane = threadIdx.x & 31;
    const int wv = threadIdx.x >> 5;
    const int bt = blockIdx.x >> 6;
    const int c  = blockIdx.x & 63;
    const float g = gatep[0];

    // async DMA: 32 bytes per thread per plane, memory -> LDS (ASYNCcnt)
    {
        const size_t base = ((size_t)(bt * 64 + c)) << 12;
        const int t = threadIdx.x * 16;
        uint32_t ldsr = (uint32_t)(uintptr_t)&sAr[0][t];
        uint32_t ldsi = (uint32_t)(uintptr_t)&sAi[0][t];
        const h16* gr = xnr + base + t;
        const h16* gi = xni + base + t;
        asm volatile(
            "global_load_async_to_lds_b128 %0, %2, off\n\t"
            "global_load_async_to_lds_b128 %0, %2, off offset:16\n\t"
            "global_load_async_to_lds_b128 %1, %3, off\n\t"
            "global_load_async_to_lds_b128 %1, %3, off offset:16\n\t"
            "s_wait_asynccnt 0"
            :: "v"(ldsr), "v"(ldsi), "v"(gr), "v"(gi) : "memory");
    }
    __syncthreads();

    for (int s = 0; s < 4; ++s) {
        const int pp = s & 1;
        const h16* Bp = (s < 2) ? fpk : fipk;
        const int m0 = (wv >> 1) << 4;            // tiles (wv*2+rep) on 4x4 grid
        v16h ar0 = load_a_frag(&sAr[pp][m0 * 64 + 0],  64);
        v16h ai0 = load_a_frag(&sAi[pp][m0 * 64 + 0],  64);
        v16h ar1 = load_a_frag(&sAr[pp][m0 * 64 + 32], 64);
        v16h ai1 = load_a_frag(&sAi[pp][m0 * 64 + 32], 64);
#pragma unroll
        for (int rep = 0; rep < 2; ++rep) {
            const int n0 = (((wv & 1) << 1) + rep) << 4;
            v8f cr = zero8(), ci = zero8();
            cgemm_pre(ar0, ai0, ar1, ai1, Bp, n0, cr, ci);
            if (s == 1) {            // twiddle with wspec (S^T indexing)
#pragma unroll
                for (int vv = 0; vv < 8; ++vv) {
                    int mg = m0 + vv + ((lane >> 4) << 3);
                    int ng = n0 + (lane & 15);
                    float wr = spec_wr[(size_t)c * 4096 + ng * 64 + mg];
                    float wi = spec_wi[(size_t)c * 4096 + ng * 64 + mg];
                    float r = cr[vv], im = ci[vv];
                    cr[vv] = r * wr - im * wi;
                    ci[vv] = r * wi + im * wr;
                }
            }
            if (s < 3) {             // transposed store into ping-pong buffer
                h16* Or = sAr[1 - pp]; h16* Oi = sAi[1 - pp];
#pragma unroll
                for (int vv = 0; vv < 8; ++vv) {
                    int mg = m0 + vv + ((lane >> 4) << 3);
                    int ng = n0 + (lane & 15);
                    Or[ng * 64 + mg] = (h16)cr[vv];
                    Oi[ng * 64 + mg] = (h16)ci[vv];
                }
            } else {                 // final: z = g*cliff + (1-g)*spec + x
#pragma unroll
                for (int vv = 0; vv < 8; ++vv) {
                    int mg = m0 + vv + ((lane >> 4) << 3);   // h
                    int ng = n0 + (lane & 15);               // w
                    int p = mg * 64 + ng;
                    float cliffr = convo[((size_t)(bt * 4096 + p) << 7) + c];
                    float cliffi = convo[((size_t)(bt * 4096 + p) << 7) + 64 + c];
                    size_t ga = ((size_t)(bt * 64 + c) << 12) + p;
                    zr[ga] = g * cliffr + (1.f - g) * cr[vv] + x_r[ga];
                    zi[ga] = g * cliffi + (1.f - g) * ci[vv] + x_i[ga];
                }
            }
        }
        __syncthreads();
    }
}

// ===========================================================================
// K5: temporal complex LayerNorm; emits zn16 [(b*HW+p)*16 + t][128]
// (16 time-rows per position = one WMMA M-tile for the SSM GEMMs)
// ===========================================================================
__global__ void k_ln_temporal(const float* zr, const float* zi,
                              const float* g, const float* bta, h16* zn16) {
    const int lane = threadIdx.x & 31;
    const int gpos = blockIdx.x * 8 + (threadIdx.x >> 5);
    const int bt = gpos >> 12, p = gpos & 4095;
    const int b = bt >> 4, t = bt & 15;
    float v[4];
#pragma unroll
    for (int j = 0; j < 4; ++j) {
        int f = lane + 32 * j;
        v[j] = (f < 64) ? zr[(size_t)((bt << 6) + f) * 4096 + p]
                        : zi[(size_t)((bt << 6) + f - 64) * 4096 + p];
    }
    float s = 0.f, s2 = 0.f;
#pragma unroll
    for (int j = 0; j < 4; ++j) { s += v[j]; s2 += v[j] * v[j]; }
    s = wave_sum(s); s2 = wave_sum(s2);
    float mu = s * (1.0f / 128.0f);
    float inv = rsqrtf(s2 * (1.0f / 128.0f) - mu * mu + 1e-5f);
    const size_t row = ((size_t)(b * 4096 + p) * 16 + t) << 7;
#pragma unroll
    for (int j = 0; j < 4; ++j) {
        int f = lane + 32 * j;
        zn16[row + f] = (h16)((v[j] - mu) * inv * g[f] + bta[f]);
    }
}

// ===========================================================================
// K6: SSM — u = zn @ B^T (WMMA), scan over T in LDS, y = h @ C^T (WMMA),
// noise + residual epilogue. One wave per position, 8 positions / block.
// ===========================================================================
__global__ void k_ssm(const h16* __restrict__ zn16, const h16* __restrict__ bpk,
                      const h16* __restrict__ cpk, const float* atab,
                      const float* dt, const float* sigma,
                      const float* zr, const float* zi,
                      float* xbr, float* xbi) {
    __shared__ h16 uh[8 * 16 * 128];                 // per-wave [16t][128] r|i
    const int lane = threadIdx.x & 31;
    const int wv = threadIdx.x >> 5;
    const int pidx = blockIdx.x * 8 + wv;            // 0..8191
    const int b = pidx >> 12, p = pidx & 4095;
    const size_t rbase = ((size_t)pidx * 16) << 7;

    // ---- u GEMM: A rows = 16 time steps, K = 64 channels (complex) ----
    {
        v16h azr0 = load_a_frag(zn16 + rbase + 0,  128);
        v16h azr1 = load_a_frag(zn16 + rbase + 32, 128);
        v16h azi0 = load_a_frag(zn16 + rbase + 64, 128);
        v16h azi1 = load_a_frag(zn16 + rbase + 96, 128);
#pragma unroll
        for (int nt = 0; nt < 4; ++nt) {
            const int n0 = nt << 4;
            v8f ur = zero8(), ui = zero8();
            cgemm_pre(azr0, azi0, azr1, azi1, bpk, n0, ur, ui);
#pragma unroll
            for (int vv = 0; vv < 8; ++vv) {
                int t = vv + ((lane >> 4) << 3);
                int ch = n0 + (lane & 15);
                uh[((wv * 16 + t) << 7) + ch]      = (h16)ur[vv];
                uh[((wv * 16 + t) << 7) + 64 + ch] = (h16)ui[vv];
            }
        }
    }
    __syncthreads();

    // ---- sequential scan h_t = a_t h_{t-1} + u_t (512 chains / block) ----
#pragma unroll
    for (int rep = 0; rep < 2; ++rep) {
        const int idx = threadIdx.x + 256 * rep;
        const int lpos = idx >> 6, ch = idx & 63;
        const int b2 = (blockIdx.x * 8 + lpos) >> 12;
        float hr = 0.f, hi = 0.f;
#pragma unroll
        for (int t = 0; t < 16; ++t) {
            const int ad = ((lpos * 16 + t) << 7) + ch;
            float ar = atab[2 * (((b2 << 4) + t) * 64 + ch) + 0];
            float ai = atab[2 * (((b2 << 4) + t) * 64 + ch) + 1];
            float u_r = (float)uh[ad], u_i = (float)uh[ad + 64];
            float nhr = ar * hr - ai * hi + u_r;
            float nhi = ar * hi + ai * hr + u_i;
            hr = nhr; hi = nhi;
            uh[ad] = (h16)hr; uh[ad + 64] = (h16)hi;
        }
    }
    __syncthreads();

    // ---- y GEMM + noise + residual ----
    {
        const h16* hb = uh + ((wv * 16) << 7);
        v16h ahr0 = load_a_frag(hb + 0,  128);
        v16h ahr1 = load_a_frag(hb + 32, 128);
        v16h ahi0 = load_a_frag(hb + 64, 128);
        v16h ahi1 = load_a_frag(hb + 96, 128);
#pragma unroll
        for (int nt = 0; nt < 4; ++nt) {
            const int n0 = nt << 4;
            v8f yr = zero8(), yi = zero8();
            cgemm_pre(ahr0, ahi0, ahr1, ahi1, cpk, n0, yr, yi);
#pragma unroll
            for (int vv = 0; vv < 8; ++vv) {
                int t = vv + ((lane >> 4) << 3);
                int ch = n0 + (lane & 15);
                int bt = (b << 4) + t;
                float sq = sigma[0] * sqrtf(dt[bt]);
                uint32_t id = (((uint32_t)(bt * 64 + ch)) << 12) | (uint32_t)p;
                float u1 = ((float)(ihash(2u * id + 1u) >> 9) + 0.5f) * (1.0f / 8388608.0f);
                float u2 = ((float)(ihash(2u * id + 2u) >> 9) + 0.5f) * (1.0f / 8388608.0f);
                float rad = sqrtf(-2.0f * __logf(u1));
                float nr = rad * __cosf(6.28318530718f * u2);
                float ni = rad * __sinf(6.28318530718f * u2);
                size_t ga = ((size_t)(bt * 64 + ch) << 12) + p;
                xbr[ga] = yr[vv] + sq * nr + zr[ga];
                xbi[ga] = yi[vv] + sq * ni + zi[ga];
            }
        }
    }
}

// ===========================================================================
// K7a: MLP LayerNorm -> lnm16 [bt*HW + p][128]
// ===========================================================================
__global__ void k_ln_mlp(const float* xr, const float* xi,
                         const float* g, const float* bta, h16* lnm16) {
    const int lane = threadIdx.x & 31;
    const int gpos = blockIdx.x * 8 + (threadIdx.x >> 5);
    const int bt = gpos >> 12, p = gpos & 4095;
    float v[4];
#pragma unroll
    for (int j = 0; j < 4; ++j) {
        int f = lane + 32 * j;
        v[j] = (f < 64) ? xr[(size_t)((bt << 6) + f) * 4096 + p]
                        : xi[(size_t)((bt << 6) + f - 64) * 4096 + p];
    }
    float s = 0.f, s2 = 0.f;
#pragma unroll
    for (int j = 0; j < 4; ++j) { s += v[j]; s2 += v[j] * v[j]; }
    s = wave_sum(s); s2 = wave_sum(s2);
    float mu = s * (1.0f / 128.0f);
    float inv = rsqrtf(s2 * (1.0f / 128.0f) - mu * mu + 1e-5f);
#pragma unroll
    for (int j = 0; j < 4; ++j) {
        int f = lane + 32 * j;
        lnm16[(size_t)gpos * 128 + f] = (h16)((v[j] - mu) * inv * g[f] + bta[f]);
    }
}

// ===========================================================================
// K7b: fused gated MLP: [16,128]@[128,1024] -> a*silu(gt) -> @[512,128]
// + bias + residual + concat(real,imag) output. LDS-resident per block.
// ===========================================================================
__global__ void k_mlp(const h16* __restrict__ lnm16, const h16* __restrict__ win,
                      const float* b_in, const h16* __restrict__ wout,
                      const float* b_out, const float* xbr, const float* xbi,
                      float* out) {
    __shared__ h16 s_up[16 * 1024];
    __shared__ h16 s_m[16 * 512];
    const int lane = threadIdx.x & 31;
    const int wv = threadIdx.x >> 5;
    const size_t row0 = (size_t)blockIdx.x * 16;
    const int bt = (int)(row0 >> 12);
    const int p0 = (int)(row0 & 4095);

    // GEMM1: hoist the 4 A fragments (K=128), N=1024 split across 8 waves
    {
        v16h af[4];
#pragma unroll
        for (int kk = 0; kk < 4; ++kk)
            af[kk] = load_a_frag(lnm16 + (row0 << 7) + kk * 32, 128);
#pragma unroll
        for (int nt = 0; nt < 8; ++nt) {
            const int n0 = wv * 128 + nt * 16;
            v8f acc = zero8();
            __builtin_prefetch(win + (size_t)(n0 + 16) * 128, 0, 0);
#pragma unroll
            for (int kk = 0; kk < 4; ++kk) {
                v16h b = load_b_frag(win + (size_t)n0 * 128 + kk * 32, 128);
                acc = wmma16(af[kk], b, acc);
            }
            const int n = n0 + (lane & 15);
            const float bias = b_in[n];
#pragma unroll
            for (int vv = 0; vv < 8; ++vv) {
                int m = vv + ((lane >> 4) << 3);
                s_up[m * 1024 + n] = (h16)(acc[vv] + bias);
            }
        }
    }
    __syncthreads();

    // a * silu(gt): 8192 elements, 32 per thread
#pragma unroll
    for (int rep = 0; rep < 32; ++rep) {
        const int idx = threadIdx.x + 256 * rep;
        const int m = idx >> 9, n = idx & 511;
        float a = (float)s_up[m * 1024 + n];
        float gt = (float)s_up[m * 1024 + 512 + n];
        s_m[m * 512 + n] = (h16)(a * gt / (1.0f + expf(-gt)));
    }
    __syncthreads();

    // GEMM2: N=128 (one 16-tile per wave), K=512 from LDS
    {
        const int n0 = wv << 4;
        v8f acc = zero8();
#pragma unroll
        for (int k0 = 0; k0 < 512; k0 += 32) {
            v16h a = load_a_frag(s_m + k0, 512);
            v16h b = load_b_frag(wout + (size_t)n0 * 512 + k0, 512);
            acc = wmma16(a, b, acc);
        }
        const int n = n0 + (lane & 15);
        const float bias = b_out[n];
#pragma unroll
        for (int vv = 0; vv < 8; ++vv) {
            int m = vv + ((lane >> 4) << 3);
            int p = p0 + m;
            float res = (n < 64) ? xbr[((size_t)(bt * 64 + n) << 12) + p]
                                 : xbi[((size_t)(bt * 64 + (n - 64)) << 12) + p];
            out[((size_t)(bt * 128 + n) << 12) + p] = acc[vv] + bias + res;
        }
    }
}

// ===========================================================================
extern "C" void kernel_launch(void* const* d_in, const int* in_sizes, int n_in,
                              void* d_out, int out_size, void* d_ws, size_t ws_size,
                              hipStream_t stream) {
    const float* x_r      = (const float*)d_in[0];
    const float* x_i      = (const float*)d_in[1];
    const float* dt       = (const float*)d_in[2];
    const float* ln_sp_g  = (const float*)d_in[3];
    const float* ln_sp_b  = (const float*)d_in[4];
    const float* conv_w   = (const float*)d_in[5];
    const float* conv_b   = (const float*)d_in[6];
    const float* spec_wr  = (const float*)d_in[7];
    const float* spec_wi  = (const float*)d_in[8];
    const float* gate     = (const float*)d_in[9];
    const float* ln_t_g   = (const float*)d_in[10];
    const float* ln_t_b   = (const float*)d_in[11];
    const float* alpha    = (const float*)d_in[12];
    const float* omega    = (const float*)d_in[13];
    const float* bop_r    = (const float*)d_in[14];
    const float* bop_i    = (const float*)d_in[15];
    const float* cop_r    = (const float*)d_in[16];
    const float* cop_i    = (const float*)d_in[17];
    const float* sigma    = (const float*)d_in[18];
    const float* ln_m_g   = (const float*)d_in[19];
    const float* ln_m_b   = (const float*)d_in[20];
    const float* mlp_w_in = (const float*)d_in[21];
    const float* mlp_b_in = (const float*)d_in[22];
    const float* mlp_w_out= (const float*)d_in[23];
    const float* mlp_b_out= (const float*)d_in[24];
    float* out = (float*)d_out;

    char* ws = (char*)d_ws;
    h16*   xn16p = (h16*)(ws + OFF_XN16P);
    h16*   xnr   = (h16*)(ws + OFF_XNR);
    h16*   xni   = (h16*)(ws + OFF_XNI);
    float* convo = (float*)(ws + OFF_CONVO);
    float* zr    = (float*)(ws + OFF_ZR);
    float* zi    = (float*)(ws + OFF_ZI);
    h16*   zn16  = (h16*)(ws + OFF_ZN16);
    float* xbr   = (float*)(ws + OFF_XBR);
    float* xbi   = (float*)(ws + OFF_XBI);
    h16*   lnm16 = (h16*)(ws + OFF_LNM16);
    h16*   convw = (h16*)(ws + OFF_CONVW);
    h16*   win   = (h16*)(ws + OFF_WIN);
    h16*   wout  = (h16*)(ws + OFF_WOUT);
    h16*   bpk   = (h16*)(ws + OFF_BPK);
    h16*   cpk   = (h16*)(ws + OFF_CPK);
    h16*   fpk   = (h16*)(ws + OFF_FPK);
    h16*   fipk  = (h16*)(ws + OFF_FIPK);
    float* atab  = (float*)(ws + OFF_ATAB);

    // zero the padded conv input (halo must be 0); interior overwritten by K1
    hipMemsetAsync(xn16p, 0, SZ_XN16P, stream);

    k_pack<<<1400, 256, 0, stream>>>(conv_w, mlp_w_in, mlp_w_out,
                                     bop_r, bop_i, cop_r, cop_i,
                                     alpha, omega, dt,
                                     convw, win, wout, bpk, cpk, fpk, fipk, atab);
    k_ln_spatial<<<NPOS / 8, 256, 0, stream>>>(x_r, x_i, ln_sp_g, ln_sp_b,
                                               xn16p, xnr, xni);
    k_conv<<<NPOS / 16, 256, 0, stream>>>(xn16p, convw, conv_b, convo);
    k_spec<<<BT * CC, 256, 0, stream>>>(xnr, xni, fpk, fipk, spec_wr, spec_wi,
                                        convo, x_r, x_i, gate, zr, zi);
    k_ln_temporal<<<NPOS / 8, 256, 0, stream>>>(zr, zi, ln_t_g, ln_t_b, zn16);
    k_ssm<<<NPOST / 8, 256, 0, stream>>>(zn16, bpk, cpk, atab, dt, sigma,
                                         zr, zi, xbr, xbi);
    k_ln_mlp<<<NPOS / 8, 256, 0, stream>>>(xbr, xbi, ln_m_g, ln_m_b, lnm16);
    k_mlp<<<NPOS / 16, 256, 0, stream>>>(lnm16, win, mlp_b_in, wout, mlp_b_out,
                                         xbr, xbi, out);
    (void)in_sizes; (void)n_in; (void)out_size; (void)ws_size;
}
